// MultiHeadAttention_35888746725739
// MI455X (gfx1250) — compile-verified, matrix operand order checked by reading, not run on hardware
//
#include <hip/hip_runtime.h>

// ---------------------------------------------------------------------------
// Linformer multi-head attention for MI455X (gfx1250), bf16 WMMA pipeline.
// All contractions run through v_wmma_f32_16x16x32_bf16 (wave32), with
// double-buffered LDS fed by global_load_async_to_lds_b128 (ASYNCcnt).
// ---------------------------------------------------------------------------

typedef __attribute__((ext_vector_type(16))) __bf16 v16bf;
typedef __attribute__((ext_vector_type(8)))  float  v8f;

union Frag {
  v16bf v;
  uint4 u[2];
};

#define BM 128
#define BN 128
#define BK 32
#define LDSP 40            // LDS row pitch in bf16 elements
#define TILE_E (BM * LDSP) // elements per LDS buffer

static __device__ __forceinline__ unsigned short f2bf(float f) {
  unsigned int u = __float_as_uint(f);
  u += 0x7FFFu + ((u >> 16) & 1u);      // round-to-nearest-even
  return (unsigned short)(u >> 16);
}
static __device__ __forceinline__ float bf2f(unsigned short b) {
  return __uint_as_float(((unsigned int)b) << 16);
}

// CDNA5 async global->LDS copy, 16 bytes per lane (tracked by ASYNCcnt).
static __device__ __forceinline__ void async_g2l_b128(unsigned lds_off,
                                                      const void* gaddr) {
  asm volatile("global_load_async_to_lds_b128 %0, %1, off"
               :: "v"(lds_off), "v"(gaddr)
               : "memory");
}
static __device__ __forceinline__ void wait_asynccnt0() {
  asm volatile("s_wait_asynccnt 0x0" ::: "memory");
}

// ---------------------------------------------------------------------------
// f32 -> bf16 conversion (vectorized float4 -> 4x bf16)
// ---------------------------------------------------------------------------
__global__ __launch_bounds__(256)
void cvt_f32_bf16(const float* __restrict__ in, unsigned short* __restrict__ out,
                  long long n) {
  long long i = ((long long)blockIdx.x * blockDim.x + threadIdx.x) * 4;
  if (i + 3 < n) {
    float4 f = *(const float4*)(in + i);
    unsigned int p0 = (unsigned int)f2bf(f.x) | ((unsigned int)f2bf(f.y) << 16);
    unsigned int p1 = (unsigned int)f2bf(f.z) | ((unsigned int)f2bf(f.w) << 16);
    uint2 pk; pk.x = p0; pk.y = p1;
    *(uint2*)(out + i) = pk;
  } else {
    for (; i < n; ++i) out[i] = f2bf(in[i]);
  }
}

// ---------------------------------------------------------------------------
// Batched strided bf16 GEMM: C = alpha*(A x B) + bias
//   APATH: 0 = A k-contiguous (async copy), 1 = A m-contiguous (reg+scatter)
//   BPATH: 0 = B k-contiguous (async copy), 1 = B n-contiguous (reg+scatter)
//   BIASM: 0 none, 1 over N, 2 over M.   CBF: 1 = bf16 out, 0 = f32 out.
//   batch offset per operand:  off = (z >> zs)*bs_hi + (z & ((1<<zs)-1))*bs_lo
// Block tile 128x128, K-step 32; 8 waves (2x4), each wave 64x32 = 8 WMMA tiles.
// Double-buffered LDS; tile t+1 streams in while tile t runs on WMMA.
// ---------------------------------------------------------------------------
template <int APATH, int BPATH, int BIASM, int CBF>
__global__ __launch_bounds__(256)
void wmma_gemm(const unsigned short* __restrict__ A,
               const unsigned short* __restrict__ B,
               const float* __restrict__ bias,
               void* __restrict__ Cv,
               int M, int N, int K,
               long long a_rs, long long a_cs,
               long long b_rs, long long b_cs,
               long long c_rs,
               long long a_bh, long long a_bl, int a_zs,
               long long b_bh, long long b_bl, int b_zs,
               long long c_bh, long long c_bl, int c_zs,
               float alpha) {
  __shared__ unsigned short As[2][TILE_E];
  __shared__ unsigned short Bs[2][TILE_E];

  const int tid  = threadIdx.x;
  const int lane = tid & 31;
  const int wid  = tid >> 5;
  const int wm   = wid >> 2;            // 0..1 : 64-row stripe
  const int wn   = wid & 3;             // 0..3 : 32-col stripe
  const int bm0  = blockIdx.y * BM;
  const int bn0  = blockIdx.x * BN;
  const int z    = blockIdx.z;

  A += (long long)(z >> a_zs) * a_bh + (long long)(z & ((1 << a_zs) - 1)) * a_bl;
  B += (long long)(z >> b_zs) * b_bh + (long long)(z & ((1 << b_zs) - 1)) * b_bl;
  const long long c_off =
      (long long)(z >> c_zs) * c_bh + (long long)(z & ((1 << c_zs) - 1)) * c_bl;

  // LDS byte offsets of the two buffers (async copy needs raw LDS addresses).
  const unsigned asbase = (unsigned)(unsigned long long)&As[0][0];
  const unsigned bsbase = (unsigned)(unsigned long long)&Bs[0][0];

  // ---- per-thread staging geometry ----
  // k-contiguous: thread covers row (tid>>1), 16 k at kb=(tid&1)*16
  const int rkc  = tid >> 1;
  const int kbkc = (tid & 1) * 16;
  // m/n-contiguous: thread covers k=(tid>>3), 16 rows at rb=(tid&7)*16
  const int ksc  = tid >> 3;
  const int rbsc = (tid & 7) * 16;

  const unsigned short* aSrc;
  long long aStep;
  unsigned aLds;
  if (APATH == 0) {
    aSrc  = A + (long long)(bm0 + rkc) * a_rs + kbkc;
    aStep = BK;
    aLds  = (unsigned)(rkc * LDSP + kbkc) * 2u;
  } else {
    aSrc  = A + (long long)ksc * a_cs + (bm0 + rbsc);
    aStep = (long long)BK * a_cs;
    aLds  = 0;
  }
  const unsigned short* bSrc;
  long long bStep;
  unsigned bLds;
  if (BPATH == 0) {
    bSrc  = B + (long long)(bn0 + rkc) * b_cs + kbkc;
    bStep = BK;
    bLds  = (unsigned)(rkc * LDSP + kbkc) * 2u;
  } else {
    bSrc  = B + (long long)ksc * b_rs + (bn0 + rbsc);
    bStep = (long long)BK * b_rs;
    bLds  = 0;
  }

  const v8f vzero = {0.f, 0.f, 0.f, 0.f, 0.f, 0.f, 0.f, 0.f};
  v8f acc[4][2];
#pragma unroll
  for (int i = 0; i < 4; ++i)
#pragma unroll
    for (int j = 0; j < 2; ++j) acc[i][j] = vzero;

  const int T = K / BK;
  uint4 a0, a1, b0, b1;   // staging regs for scatter paths

  // ---- prologue: tile 0 into buffer 0 ----
  if (APATH == 0) {
    async_g2l_b128(asbase + aLds,      aSrc);
    async_g2l_b128(asbase + aLds + 16, aSrc + 8);
  } else {
    a0 = *(const uint4*)(aSrc);
    a1 = *(const uint4*)(aSrc + 8);
    const unsigned short* s0 = (const unsigned short*)&a0;
    const unsigned short* s1 = (const unsigned short*)&a1;
#pragma unroll
    for (int i = 0; i < 8; ++i) As[0][(rbsc + i) * LDSP + ksc]     = s0[i];
#pragma unroll
    for (int i = 0; i < 8; ++i) As[0][(rbsc + 8 + i) * LDSP + ksc] = s1[i];
  }
  if (BPATH == 0) {
    async_g2l_b128(bsbase + bLds,      bSrc);
    async_g2l_b128(bsbase + bLds + 16, bSrc + 8);
  } else {
    b0 = *(const uint4*)(bSrc);
    b1 = *(const uint4*)(bSrc + 8);
    const unsigned short* s0 = (const unsigned short*)&b0;
    const unsigned short* s1 = (const unsigned short*)&b1;
#pragma unroll
    for (int i = 0; i < 8; ++i) Bs[0][(rbsc + i) * LDSP + ksc]     = s0[i];
#pragma unroll
    for (int i = 0; i < 8; ++i) Bs[0][(rbsc + 8 + i) * LDSP + ksc] = s1[i];
  }

  const int koff = (lane & 16) ? 8 : 0;
  const int rsel = lane & 15;
  int cur = 0;

  for (int t = 0; t < T; ++t) {
    if (APATH == 0 || BPATH == 0) wait_asynccnt0();
    __syncthreads();                   // tile t resident in buf[cur]

    const bool havenext = (t + 1 < T);
    const int nxt = cur ^ 1;
    if (havenext) {
      // stream tile t+1 into buf[nxt]; overlaps with WMMA below
      if (APATH == 0) {
        const unsigned short* s = aSrc + (long long)(t + 1) * aStep;
        unsigned d = asbase + (unsigned)nxt * (TILE_E * 2u) + aLds;
        async_g2l_b128(d, s);
        async_g2l_b128(d + 16, s + 8);
      } else {
        const unsigned short* s = aSrc + (long long)(t + 1) * aStep;
        a0 = *(const uint4*)(s);
        a1 = *(const uint4*)(s + 8);
      }
      if (BPATH == 0) {
        const unsigned short* s = bSrc + (long long)(t + 1) * bStep;
        unsigned d = bsbase + (unsigned)nxt * (TILE_E * 2u) + bLds;
        async_g2l_b128(d, s);
        async_g2l_b128(d + 16, s + 8);
      } else {
        const unsigned short* s = bSrc + (long long)(t + 1) * bStep;
        b0 = *(const uint4*)(s);
        b1 = *(const uint4*)(s + 8);
      }
    }

    // ---- WMMA on tile t ----
    Frag af[4], bf[2];
#pragma unroll
    for (int i = 0; i < 4; ++i) {
      const unsigned short* p = &As[cur][(wm * 64 + i * 16 + rsel) * LDSP + koff];
      af[i].u[0] = *(const uint4*)(p);
      af[i].u[1] = *(const uint4*)(p + 16);
    }
#pragma unroll
    for (int j = 0; j < 2; ++j) {
      const unsigned short* p = &Bs[cur][(wn * 32 + j * 16 + rsel) * LDSP + koff];
      bf[j].u[0] = *(const uint4*)(p);
      bf[j].u[1] = *(const uint4*)(p + 16);
    }
#pragma unroll
    for (int i = 0; i < 4; ++i)
#pragma unroll
      for (int j = 0; j < 2; ++j)
        acc[i][j] = __builtin_amdgcn_wmma_f32_16x16x32_bf16(
            false, af[i].v, false, bf[j].v, (short)0, acc[i][j], false, false);

    // ---- deferred scatter stores for tile t+1 (load latency hidden) ----
    if (havenext) {
      if (APATH == 1) {
        const unsigned short* s0 = (const unsigned short*)&a0;
        const unsigned short* s1 = (const unsigned short*)&a1;
#pragma unroll
        for (int i = 0; i < 8; ++i) As[nxt][(rbsc + i) * LDSP + ksc]     = s0[i];
#pragma unroll
        for (int i = 0; i < 8; ++i) As[nxt][(rbsc + 8 + i) * LDSP + ksc] = s1[i];
      }
      if (BPATH == 1) {
        const unsigned short* s0 = (const unsigned short*)&b0;
        const unsigned short* s1 = (const unsigned short*)&b1;
#pragma unroll
        for (int i = 0; i < 8; ++i) Bs[nxt][(rbsc + i) * LDSP + ksc]     = s0[i];
#pragma unroll
        for (int i = 0; i < 8; ++i) Bs[nxt][(rbsc + 8 + i) * LDSP + ksc] = s1[i];
      }
    }
    cur = nxt;
  }

  // ---- epilogue: scale + bias + store (f32 or bf16) ----
  const int mhalf = (lane & 16) ? 8 : 0;
#pragma unroll
  for (int i = 0; i < 4; ++i) {
#pragma unroll
    for (int j = 0; j < 2; ++j) {
      const int gn = bn0 + wn * 32 + j * 16 + (lane & 15);
      float badd = (BIASM == 1) ? bias[gn] : 0.0f;
#pragma unroll
      for (int r = 0; r < 8; ++r) {
        const int gm = bm0 + wm * 64 + i * 16 + mhalf + r;
        float v = acc[i][j][r] * alpha + badd;
        if (BIASM == 2) v += bias[gm];
        const long long idx = c_off + (long long)gm * c_rs + gn;
        if (CBF) ((unsigned short*)Cv)[idx] = f2bf(v);
        else     ((float*)Cv)[idx]          = v;
      }
    }
  }
}

// ---------------------------------------------------------------------------
// In-place softmax over rows of 256 bf16 (one wave per row, 8 elems/lane).
// row = (b*8 + h)*4096 + s ; masked query rows -> uniform 1/256.
// ---------------------------------------------------------------------------
__global__ __launch_bounds__(256)
void softmax256_bf16(unsigned short* __restrict__ attn,
                     const unsigned char* __restrict__ mask) {
  const int row  = blockIdx.x * 8 + (threadIdx.x >> 5);
  const int lane = threadIdx.x & 31;
  const int b = row >> 15;
  const int s = row & 4095;
  unsigned short* p = attn + (long long)row * 256 + lane * 8;
  uint4 d = *(uint4*)p;
  unsigned short* e = (unsigned short*)&d;
  float x[8];
#pragma unroll
  for (int i = 0; i < 8; ++i) x[i] = bf2f(e[i]);

  float mx = x[0];
#pragma unroll
  for (int i = 1; i < 8; ++i) mx = fmaxf(mx, x[i]);
#pragma unroll
  for (int off = 16; off > 0; off >>= 1) mx = fmaxf(mx, __shfl_xor(mx, off, 32));

  float sum = 0.0f;
#pragma unroll
  for (int i = 0; i < 8; ++i) { x[i] = __expf(x[i] - mx); sum += x[i]; }
#pragma unroll
  for (int off = 16; off > 0; off >>= 1) sum += __shfl_xor(sum, off, 32);

  const bool valid = mask[b * 4096 + s] != 0;
  const float inv = valid ? (1.0f / sum) : 0.0f;
  const float uni = 1.0f / 256.0f;
#pragma unroll
  for (int i = 0; i < 8; ++i) e[i] = f2bf(valid ? x[i] * inv : uni);
  *(uint4*)p = d;
}

// ---------------------------------------------------------------------------
// Host orchestration
// ---------------------------------------------------------------------------
extern "C" void kernel_launch(void* const* d_in, const int* in_sizes, int n_in,
                              void* d_out, int out_size, void* d_ws, size_t ws_size,
                              hipStream_t stream) {
  (void)in_sizes; (void)n_in; (void)out_size; (void)ws_size;

  constexpr int B = 4, S = 4096, DM = 512, H = 8, HD = 4096, DK = 256;
  constexpr long long MS  = (long long)B * S;        // 16384 rows
  constexpr long long SHD = (long long)S * HD;       // per-batch plane (elems)

  const float* qf = (const float*)d_in[0];
  const float* kf = (const float*)d_in[1];
  const float* vf = (const float*)d_in[2];
  const unsigned char* mask = (const unsigned char*)d_in[3];
  const float* Wq = (const float*)d_in[4];  const float* bq = (const float*)d_in[5];
  const float* Wk = (const float*)d_in[6];  const float* bk = (const float*)d_in[7];
  const float* Wv = (const float*)d_in[8];  const float* bv = (const float*)d_in[9];
  const float* Ew = (const float*)d_in[10]; const float* Eb = (const float*)d_in[11];
  const float* Fw = (const float*)d_in[12]; const float* Fb = (const float*)d_in[13];
  const float* Wo = (const float*)d_in[14]; const float* bo = (const float*)d_in[15];
  float* out = (float*)d_out;

  char* ws = (char*)d_ws;
  size_t off = 0;
  auto alloc = [&](long long elems) {
    off = (off + 255) & ~(size_t)255;
    unsigned short* p = (unsigned short*)(ws + off);
    off += (size_t)elems * 2;
    return p;
  };
  unsigned short* Wqb = alloc((long long)HD * DM);
  unsigned short* Wkb = alloc((long long)HD * DM);
  unsigned short* Wvb = alloc((long long)HD * DM);
  unsigned short* Wob = alloc((long long)DM * HD);
  unsigned short* Ewb = alloc((long long)DK * S);
  unsigned short* Fwb = alloc((long long)DK * S);
  unsigned short* qxb = alloc(MS * DM);
  unsigned short* kxb = alloc(MS * DM);
  unsigned short* vxb = alloc(MS * DM);
  unsigned short* Yb  = alloc(MS * HD);               // reused: kY, vY, qY, x
  unsigned short* kp  = alloc((long long)B * H * DM * DK);
  unsigned short* vp  = alloc((long long)B * H * DK * DM);
  unsigned short* at  = alloc((long long)B * H * S * DK);

  auto cvt = [&](const float* src, unsigned short* dst, long long n) {
    long long vec = (n + 3) / 4;
    int blocks = (int)((vec + 255) / 256);
    cvt_f32_bf16<<<blocks, 256, 0, stream>>>(src, dst, n);
  };

  // 1) convert weights + activations to bf16
  cvt(Wq, Wqb, (long long)HD * DM);
  cvt(Wk, Wkb, (long long)HD * DM);
  cvt(Wv, Wvb, (long long)HD * DM);
  cvt(Wo, Wob, (long long)DM * HD);
  cvt(Ew, Ewb, (long long)DK * S);
  cvt(Fw, Fwb, (long long)DK * S);
  cvt(qf, qxb, MS * DM);
  cvt(kf, kxb, MS * DM);
  cvt(vf, vxb, MS * DM);

  const float inv_sqrt_d = 0.04419417382415922f;  // 1/sqrt(512)
  const dim3 blk(256);

  // 2) kY = key @ Wk^T + bk -> Yb    [A kcontig, B kcontig, biasN, bf16]
  wmma_gemm<0,0,1,1><<<dim3(HD/BN, MS/BM, 1), blk, 0, stream>>>(
      kxb, Wkb, bk, Yb, (int)MS, HD, DM,
      DM, 1, 1, DM, HD, 0, 0, 0, 0, 0, 0, 0, 0, 0, 1.0f);
  // 3) kp[z][d][r] = sum_s kY[(b*S+s)*HD + h*512 + d]*Ew[r*S+s] + Eb[r]
  //    [A mcontig, B kcontig, biasN, bf16]
  wmma_gemm<1,0,1,1><<<dim3(DK/BN, DM/BM, B*H), blk, 0, stream>>>(
      Yb, Ewb, Eb, kp, DM, DK, S,
      1, HD, 1, S, DK,
      SHD, DM, 3, 0, 0, 0, (long long)DM * DK, 0, 0, 1.0f);
  // 4) vY = value @ Wv^T + bv -> Yb
  wmma_gemm<0,0,1,1><<<dim3(HD/BN, MS/BM, 1), blk, 0, stream>>>(
      vxb, Wvb, bv, Yb, (int)MS, HD, DM,
      DM, 1, 1, DM, HD, 0, 0, 0, 0, 0, 0, 0, 0, 0, 1.0f);
  // 5) vp[z][r][d] = sum_s Fw[r*S+s]*vY[(b*S+s)*HD + h*512 + d] + Fb[r]
  //    [A kcontig, B ncontig, biasM, bf16]
  wmma_gemm<0,1,2,1><<<dim3(DM/BN, DK/BM, B*H), blk, 0, stream>>>(
      Fwb, Yb, Fb, vp, DK, DM, S,
      S, 1, HD, 1, DM,
      0, 0, 0, SHD, DM, 3, (long long)DK * DM, 0, 0, 1.0f);
  // 6) qY = query @ Wq^T + bq -> Yb
  wmma_gemm<0,0,1,1><<<dim3(HD/BN, MS/BM, 1), blk, 0, stream>>>(
      qxb, Wqb, bq, Yb, (int)MS, HD, DM,
      DM, 1, 1, DM, HD, 0, 0, 0, 0, 0, 0, 0, 0, 0, 1.0f);
  // 7) logits[z][s][r] = (qY . kp) / sqrt(512) -> at
  //    [A kcontig, B ncontig, no bias, bf16]
  wmma_gemm<0,1,0,1><<<dim3(DK/BN, S/BM, B*H), blk, 0, stream>>>(
      Yb, kp, nullptr, at, S, DK, DM,
      HD, 1, DK, 1, DK,
      SHD, DM, 3, (long long)DM * DK, 0, 0, (long long)S * DK, 0, 0, inv_sqrt_d);
  // 8) softmax rows of 256 in place
  softmax256_bf16<<<(B * H * S) / 8, blk, 0, stream>>>(at, mask);
  // 9) x[(b*S+s)*HD + h*512 + d] = sum_r at[z][s][r]*vp[z][r][d] -> Yb
  wmma_gemm<0,1,0,1><<<dim3(DM/BN, S/BM, B*H), blk, 0, stream>>>(
      at, vp, nullptr, Yb, S, DM, DK,
      DK, 1, DM, 1, HD,
      (long long)S * DK, 0, 0, (long long)DK * DM, 0, 0, SHD, DM, 3, 1.0f);
  // 10) out = x @ Wo^T + bo  (f32 output)
  wmma_gemm<0,0,1,0><<<dim3(DM/BN, MS/BM, 1), blk, 0, stream>>>(
      Yb, Wob, bo, out, (int)MS, DM, HD,
      HD, 1, 1, HD, DM,
      0, 0, 0, 0, 0, 0, 0, 0, 0, 1.0f);
}
